// SplineDQN_90649579749935
// MI455X (gfx1250) — compile-verified, matrix operand order checked by reading, not run on hardware
//
#include <hip/hip_runtime.h>
#include <hip/hip_bf16.h>

typedef __attribute__((ext_vector_type(16))) __bf16 v16bf;
typedef __attribute__((ext_vector_type(8)))  float  v8f;

#define BDIM 16384
#define NUM_INPUTS 256
#define ACTION_DIM 16
#define H0 1024
#define H1 1024
#define KBINS 128
#define K2PAD 1056          // 1040 padded to multiple of 32
#define N3PAD 400           // 383 + wa + wb, padded to multiple of 16

// ---- fp32 -> bf16 (round to nearest even) ----
static __device__ __forceinline__ unsigned short f32_to_bf16(float f) {
    unsigned int u = __float_as_uint(f);
    unsigned int lsb = (u >> 16) & 1u;
    u += 0x7fffu + lsb;
    return (unsigned short)(u >> 16);
}

union FragBF {
    v16bf v;
    uint4 q[2];
};

// ---- CDNA5 async global->LDS copy (ASYNCcnt-tracked DMA, no VGPR staging) ----
static __device__ __forceinline__ void async_copy_b128(const void* gaddr,
                                                       unsigned lds_off) {
    asm volatile("global_load_async_to_lds_b128 %0, %1, off"
                 :: "v"(lds_off), "v"(gaddr)
                 : "memory");
}
#define WAIT_ASYNCCNT(n) asm volatile("s_wait_asynccnt %0" :: "i"(n) : "memory")

// generic __shared__ pointer -> raw LDS byte offset (ISA 10.2: addr[31:0])
static __device__ __forceinline__ unsigned lds_off_of(const void* p) {
    return (unsigned)(unsigned long long)p;
}

// =====================================================================
// Conversion / packing kernels
// =====================================================================
__global__ void cvt_bf16_kernel(const float* __restrict__ src,
                                unsigned short* __restrict__ dst, int n) {
    int i = blockIdx.x * 256 + threadIdx.x;
    if (i < n) dst[i] = f32_to_bf16(src[i]);
}

// w2 is [1024 x 1040] -> dst [1024 x 1056], zero pad cols 1040..1055
__global__ void cvt_pad_w2_kernel(const float* __restrict__ w2,
                                  unsigned short* __restrict__ dst) {
    int i = blockIdx.x * 256 + threadIdx.x;
    if (i >= H1 * K2PAD) return;
    int r = i / K2PAD, c = i % K2PAD;
    dst[i] = (c < (H0 + ACTION_DIM)) ? f32_to_bf16(w2[r * (H0 + ACTION_DIM) + c])
                                     : (unsigned short)0;
}

// build [400 x 1024]: rows 0..382 = wv, 383 = wa, 384 = wb, rest zero
__global__ void build_wvb_kernel(const float* __restrict__ wv,
                                 const float* __restrict__ wa,
                                 const float* __restrict__ wb,
                                 unsigned short* __restrict__ dst) {
    int i = blockIdx.x * 256 + threadIdx.x;
    if (i >= N3PAD * H1) return;
    int r = i / H1, c = i % H1;
    float x = 0.0f;
    if (r < 3 * KBINS - 1)       x = wv[r * H1 + c];
    else if (r == 3 * KBINS - 1) x = wa[c];
    else if (r == 3 * KBINS)     x = wb[c];
    dst[i] = f32_to_bf16(x);
}

// =====================================================================
// WMMA GEMM:  C[M x N] = A[M x Kd] * Bw[N x Kd]^T   (bf16 in, f32 out)
// BM=128, BN=128, BK=32; 256 threads = 8 waves; wave = 64x32 = 4x2 WMMAs.
// Double-buffered LDS filled by global_load_async_to_lds_b128.
// Pipeline is unrolled by two tiles with a peeled epilogue so all buffer
// indices are compile-time constants (no per-iteration branches/selects).
// =====================================================================
#define LDS_STRIDE 40   // 32 + 8 shorts pad; 80 bytes per row (16B aligned)
#define TILE_SH (128 * LDS_STRIDE)

__global__ __launch_bounds__(256)
void gemm_bf16_wmma(const unsigned short* __restrict__ A,
                    const unsigned short* __restrict__ Bw,
                    float* __restrict__ C,
                    int N, int Kd, int lda, int ldb, int ldc) {
    __shared__ __align__(16) unsigned short sA[2][TILE_SH];
    __shared__ __align__(16) unsigned short sB[2][TILE_SH];

    const int tid  = threadIdx.x;
    const int lane = tid & 31;
    const int wave = tid >> 5;
    const int wm   = wave >> 2;      // 0..1
    const int wn   = wave & 3;       // 0..3
    const int hl   = lane >> 4;      // 0..1
    const int l16  = lane & 15;      // 0..15

    const int rowBase = blockIdx.y * 128;
    const int colBase = blockIdx.x * 128;

    // per-thread copy chunks: tile = 512 x 16B chunks, 2 A + 2 B per thread
    const int c0 = tid, c1 = tid + 256;
    const int r0 = c0 >> 2, kc0 = (c0 & 3) * 8;
    const int r1 = c1 >> 2, kc1 = (c1 & 3) * 8;
    // clamp B row: OOB rows feed only accumulators of cols >= N (never stored)
    const int nn0 = (colBase + r0 < N) ? (colBase + r0) : (N - 1);
    const int nn1 = (colBase + r1 < N) ? (colBase + r1) : (N - 1);

    const unsigned short* ga0 = &A[(size_t)(rowBase + r0) * lda + kc0];
    const unsigned short* ga1 = &A[(size_t)(rowBase + r1) * lda + kc1];
    const unsigned short* gb0 = &Bw[(size_t)nn0 * ldb + kc0];
    const unsigned short* gb1 = &Bw[(size_t)nn1 * ldb + kc1];

    unsigned la0[2], la1[2], lb0[2], lb1[2];
    #pragma unroll
    for (int b = 0; b < 2; ++b) {
        la0[b] = lds_off_of(&sA[b][r0 * LDS_STRIDE + kc0]);
        la1[b] = lds_off_of(&sA[b][r1 * LDS_STRIDE + kc1]);
        lb0[b] = lds_off_of(&sB[b][r0 * LDS_STRIDE + kc0]);
        lb1[b] = lds_off_of(&sB[b][r1 * LDS_STRIDE + kc1]);
    }

    v8f acc[4][2];
    #pragma unroll
    for (int mt = 0; mt < 4; ++mt)
        #pragma unroll
        for (int nt = 0; nt < 2; ++nt)
            #pragma unroll
            for (int r = 0; r < 8; ++r) acc[mt][nt][r] = 0.0f;

    // issue tile at K-offset kk into buffer b (4 async b128 chunks / thread)
    auto issue = [&](int kk, int b) {
        async_copy_b128(ga0 + kk, la0[b]);
        async_copy_b128(ga1 + kk, la1[b]);
        async_copy_b128(gb0 + kk, lb0[b]);
        async_copy_b128(gb1 + kk, lb1[b]);
    };

    // 8 WMMAs on the tile in buffer b
    auto compute = [&](int b) {
        const unsigned short* tA = sA[b];
        const unsigned short* tB = sB[b];
        FragBF bfr[2];
        #pragma unroll
        for (int nt = 0; nt < 2; ++nt) {
            int r = wn * 32 + nt * 16 + l16;
            bfr[nt].q[0] = *(const uint4*)&tB[r * LDS_STRIDE + 8 * hl];
            bfr[nt].q[1] = *(const uint4*)&tB[r * LDS_STRIDE + 16 + 8 * hl];
        }
        #pragma unroll
        for (int mt = 0; mt < 4; ++mt) {
            FragBF af;
            int r = wm * 64 + mt * 16 + l16;
            af.q[0] = *(const uint4*)&tA[r * LDS_STRIDE + 8 * hl];
            af.q[1] = *(const uint4*)&tA[r * LDS_STRIDE + 16 + 8 * hl];
            #pragma unroll
            for (int nt = 0; nt < 2; ++nt) {
                acc[mt][nt] = __builtin_amdgcn_wmma_f32_16x16x32_bf16(
                    false, af.v, false, bfr[nt].v, (short)0, acc[mt][nt],
                    false, false);
            }
        }
    };

    // ---- software pipeline: tile k always resident before compute ----
    issue(0, 0);
    int k = 0;
    for (; k + 64 < Kd; k += 64) {
        issue(k + 32, 1); WAIT_ASYNCCNT(4); __syncthreads();
        compute(0);       __syncthreads();
        issue(k + 64, 0); WAIT_ASYNCCNT(4); __syncthreads();
        compute(1);       __syncthreads();
    }
    if (k + 32 < Kd) {          // two tiles remain: k (buf0), k+32 (buf1)
        issue(k + 32, 1); WAIT_ASYNCCNT(4); __syncthreads();
        compute(0);       __syncthreads();
        WAIT_ASYNCCNT(0); __syncthreads();
        compute(1);
    } else {                    // one tile remains: k (buf0)
        WAIT_ASYNCCNT(0); __syncthreads();
        compute(0);
    }

    // store (C/D layout: VGPR r -> M = r + 8*half, N = lane&15)
    #pragma unroll
    for (int mt = 0; mt < 4; ++mt) {
        #pragma unroll
        for (int nt = 0; nt < 2; ++nt) {
            int col = colBase + wn * 32 + nt * 16 + l16;
            if (col < N) {
                #pragma unroll
                for (int r = 0; r < 8; ++r) {
                    int row = rowBase + wm * 64 + mt * 16 + hl * 8 + r;
                    C[(size_t)row * ldc + col] = acc[mt][nt][r];
                }
            }
        }
    }
}

// =====================================================================
// Fused bias + LayerNorm + ReLU -> bf16; optionally append bf16 actions
// + zero pad (for x2 input of GEMM2). One block (256 thr) per row of 1024.
// =====================================================================
__global__ __launch_bounds__(256)
void ln_relu_kernel(const float* __restrict__ y,
                    const float* __restrict__ bias,
                    const float* __restrict__ gamma,
                    const float* __restrict__ beta,
                    const float* __restrict__ actions,   // may be null
                    unsigned short* __restrict__ out, int ldo) {
    __shared__ float red[256];
    const int m = blockIdx.x;
    const int t = threadIdx.x;
    const float* row = y + (size_t)m * H0;

    float v[4];
    float s = 0.0f;
    #pragma unroll
    for (int i = 0; i < 4; ++i) {
        int c = t + i * 256;
        float x = row[c] + bias[c];
        v[i] = x;
        s += x;
    }
    red[t] = s; __syncthreads();
    for (int o = 128; o > 0; o >>= 1) {
        if (t < o) red[t] += red[t + o];
        __syncthreads();
    }
    float mu = red[0] * (1.0f / 1024.0f);
    __syncthreads();

    float s2 = 0.0f;
    #pragma unroll
    for (int i = 0; i < 4; ++i) { float d = v[i] - mu; s2 += d * d; }
    red[t] = s2; __syncthreads();
    for (int o = 128; o > 0; o >>= 1) {
        if (t < o) red[t] += red[t + o];
        __syncthreads();
    }
    float var = red[0] * (1.0f / 1024.0f);
    float rstd = rsqrtf(var + 1e-5f);

    unsigned short* orow = out + (size_t)m * ldo;
    #pragma unroll
    for (int i = 0; i < 4; ++i) {
        int c = t + i * 256;
        float r = (v[i] - mu) * rstd * gamma[c] + beta[c];
        orow[c] = f32_to_bf16(fmaxf(r, 0.0f));
    }
    if (actions != nullptr) {
        if (t < ACTION_DIM)
            orow[H0 + t] = f32_to_bf16(actions[(size_t)m * ACTION_DIM + t]);
        else if (t < 32)
            orow[H0 + t] = 0;   // pad cols 1040..1055
    }
}

// =====================================================================
// Spline kernel: one block of 128 threads per batch row.
// =====================================================================
__global__ __launch_bounds__(128)
void spline_kernel(const float* __restrict__ sp,     // [B x 400]
                   const float* __restrict__ bv,     // [383]
                   const float* __restrict__ ba,     // [1]
                   const float* __restrict__ bb,     // [1]
                   float* __restrict__ out) {        // [B x 128]
    __shared__ float red[128];
    __shared__ float cw[KBINS + 1];
    __shared__ float ch[KBINS + 1];
    __shared__ float Dar[KBINS + 1];
    __shared__ float sAB[2];

    const int m = blockIdx.x;
    const int t = threadIdx.x;
    const float* row = sp + (size_t)m * N3PAD;

    const float EDGE = logf(expf(1.0f - 0.001f) - 1.0f);
    const float MBW = 0.001f, MBH = 0.001f, MDER = 0.001f;

    float wraw = row[t]         + bv[t];
    float hraw = row[KBINS + t] + bv[KBINS + t];
    float draw = (t < KBINS - 1) ? (row[2 * KBINS + t] + bv[2 * KBINS + t]) : 0.0f;
    if (t == 0) {
        sAB[0] = expf(row[3 * KBINS - 1] + ba[0]);  // scale_a
        sAB[1] = row[3 * KBINS] + bb[0];            // scale_b
    }

    // ---- softmax(W) ----
    red[t] = wraw; __syncthreads();
    for (int o = 64; o > 0; o >>= 1) { if (t < o) red[t] = fmaxf(red[t], red[t + o]); __syncthreads(); }
    float wmax = red[0]; __syncthreads();
    float we = expf(wraw - wmax);
    red[t] = we; __syncthreads();
    for (int o = 64; o > 0; o >>= 1) { if (t < o) red[t] += red[t + o]; __syncthreads(); }
    float Wv = MBW + (1.0f - MBW * KBINS) * we / red[0];
    __syncthreads();

    // ---- softmax(H) ----
    red[t] = hraw; __syncthreads();
    for (int o = 64; o > 0; o >>= 1) { if (t < o) red[t] = fmaxf(red[t], red[t + o]); __syncthreads(); }
    float hmax = red[0]; __syncthreads();
    float he = expf(hraw - hmax);
    red[t] = he; __syncthreads();
    for (int o = 64; o > 0; o >>= 1) { if (t < o) red[t] += red[t + o]; __syncthreads(); }
    float Hv = MBH + (1.0f - MBH * KBINS) * he / red[0];
    __syncthreads();

    // ---- inclusive scan of W -> cumwidths ----
    red[t] = Wv; __syncthreads();
    for (int o = 1; o < KBINS; o <<= 1) {
        float add = (t >= o) ? red[t - o] : 0.0f;
        __syncthreads();
        red[t] += add;
        __syncthreads();
    }
    cw[t + 1] = (t == KBINS - 1) ? 1.0f : red[t];
    if (t == 0) cw[0] = 0.0f;
    __syncthreads();

    // ---- inclusive scan of H -> cumheights (affine with scale_a/b) ----
    red[t] = Hv; __syncthreads();
    for (int o = 1; o < KBINS; o <<= 1) {
        float add = (t >= o) ? red[t - o] : 0.0f;
        __syncthreads();
        red[t] += add;
        __syncthreads();
    }
    float sa = sAB[0], sb = sAB[1];
    ch[t + 1] = sa * red[t] + sb;
    if (t == 0) ch[0] = sb;

    // ---- derivatives: D = [edge, softplus-mids(127), edge] ----
    if (t < KBINS - 1) {
        float spv = (draw > 20.0f) ? draw : log1pf(expf(draw));
        Dar[t + 1] = MDER + spv;
    }
    if (t == 0) { Dar[0] = EDGE; Dar[KBINS] = EDGE; }
    __syncthreads();

    // ---- bin search + rational-quadratic evaluation ----
    float tau = ((float)t + 0.5f) * (1.0f / (float)KBINS);
    int lo = 0, hi = KBINS;
    #pragma unroll
    for (int it = 0; it < 7; ++it) {
        int mid = (lo + hi) >> 1;
        if (tau >= cw[mid]) lo = mid; else hi = mid;
    }
    int b = lo;

    float icw = cw[b];
    float ibw = cw[b + 1] - icw;
    float ich = ch[b];
    float ih  = ch[b + 1] - ich;
    float idel = ih / ibw;
    float d0 = Dar[b];
    float d1 = Dar[b + 1];

    float theta = (tau - icw) / ibw;
    float th1m  = theta * (1.0f - theta);
    float numer = ih * (idel * theta * theta + d0 * th1m);
    float denom = idel + (d0 + d1 - 2.0f * idel) * th1m;
    out[(size_t)m * KBINS + t] = ich + numer / denom;
}

// =====================================================================
// Host-side launch
// =====================================================================
extern "C" void kernel_launch(void* const* d_in, const int* in_sizes, int n_in,
                              void* d_out, int out_size, void* d_ws, size_t ws_size,
                              hipStream_t stream) {
    const float* inputs  = (const float*)d_in[0];
    const float* actions = (const float*)d_in[1];
    const float* w1      = (const float*)d_in[2];
    const float* b1      = (const float*)d_in[3];
    const float* ln1_g   = (const float*)d_in[4];
    const float* ln1_b   = (const float*)d_in[5];
    const float* w2      = (const float*)d_in[6];
    const float* b2      = (const float*)d_in[7];
    const float* ln2_g   = (const float*)d_in[8];
    const float* ln2_b   = (const float*)d_in[9];
    const float* wv      = (const float*)d_in[10];
    const float* bv      = (const float*)d_in[11];
    const float* wa      = (const float*)d_in[12];
    const float* ba      = (const float*)d_in[13];
    const float* wb      = (const float*)d_in[14];
    const float* bb      = (const float*)d_in[15];
    float* out = (float*)d_out;

    // workspace layout (bytes)
    char* ws = (char*)d_ws;
    const size_t OFF_W1B  = 0;                               // 1024x256  bf16
    const size_t OFF_W2B  = OFF_W1B  + (size_t)H1 * 256 * 2; // 1024x1056 bf16
    const size_t OFF_WVB  = OFF_W2B  + (size_t)H1 * K2PAD * 2;
    const size_t OFF_XB   = OFF_WVB  + (size_t)N3PAD * H1 * 2;
    const size_t OFF_X2IN = OFF_XB   + (size_t)BDIM * NUM_INPUTS * 2;
    const size_t OFF_X2   = OFF_X2IN + (size_t)BDIM * K2PAD * 2;
    const size_t OFF_Y    = OFF_X2   + (size_t)BDIM * H1 * 2;
    const size_t OFF_SP   = OFF_Y    + (size_t)BDIM * H1 * 4;

    unsigned short* w1b  = (unsigned short*)(ws + OFF_W1B);
    unsigned short* w2b  = (unsigned short*)(ws + OFF_W2B);
    unsigned short* wvb  = (unsigned short*)(ws + OFF_WVB);
    unsigned short* xb   = (unsigned short*)(ws + OFF_XB);
    unsigned short* x2in = (unsigned short*)(ws + OFF_X2IN);
    unsigned short* x2   = (unsigned short*)(ws + OFF_X2);
    float*          y    = (float*)(ws + OFF_Y);
    float*          sp   = (float*)(ws + OFF_SP);

    // 1) conversions
    {
        int n = BDIM * NUM_INPUTS;
        cvt_bf16_kernel<<<(n + 255) / 256, 256, 0, stream>>>(inputs, xb, n);
    }
    {
        int n = H0 * NUM_INPUTS;
        cvt_bf16_kernel<<<(n + 255) / 256, 256, 0, stream>>>(w1, w1b, n);
    }
    cvt_pad_w2_kernel<<<(H1 * K2PAD + 255) / 256, 256, 0, stream>>>(w2, w2b);
    build_wvb_kernel<<<(N3PAD * H1 + 255) / 256, 256, 0, stream>>>(wv, wa, wb, wvb);

    // 2) GEMM1: y = xb @ w1b^T   [16384 x 1024]
    gemm_bf16_wmma<<<dim3(H0 / 128, BDIM / 128), 256, 0, stream>>>(
        xb, w1b, y, H0, NUM_INPUTS, NUM_INPUTS, NUM_INPUTS, H0);

    // 3) LN1 + ReLU + concat actions -> x2in (bf16, 1056 wide)
    ln_relu_kernel<<<BDIM, 256, 0, stream>>>(y, b1, ln1_g, ln1_b, actions,
                                             x2in, K2PAD);

    // 4) GEMM2: y = x2in @ w2b^T  [16384 x 1024]
    gemm_bf16_wmma<<<dim3(H1 / 128, BDIM / 128), 256, 0, stream>>>(
        x2in, w2b, y, H1, K2PAD, K2PAD, K2PAD, H1);

    // 5) LN2 + ReLU -> x2 (bf16, 1024 wide)
    ln_relu_kernel<<<BDIM, 256, 0, stream>>>(y, b2, ln2_g, ln2_b, nullptr,
                                             x2, H1);

    // 6) GEMM3: sp = x2 @ wvb^T  [16384 x 400]
    gemm_bf16_wmma<<<dim3((N3PAD + 127) / 128, BDIM / 128), 256, 0, stream>>>(
        x2, wvb, sp, N3PAD, H1, H1, H1, N3PAD);

    // 7) spline evaluation -> out [16384 x 128]
    spline_kernel<<<BDIM, 128, 0, stream>>>(sp, bv, ba, bb, out);
}
